// AttentionModule_11819749998882
// MI455X (gfx1250) — matrix operand order repaired and checked by reading, not compiled, not run on hardware
//
#include <hip/hip_runtime.h>

// ---------------------------------------------------------------------------
// Fused 1x1-conv attention for MI455X (gfx1250, wave32)
//   WMMA 16x16x32 bf16 + Tensor Data Mover double-buffered K/V staging
//   B=8, C=256, HID=32, H=W=64, N=4096
// ---------------------------------------------------------------------------

typedef __attribute__((ext_vector_type(16))) __bf16 v16bf;
typedef __attribute__((ext_vector_type(8)))  __bf16 v8bf;
typedef __attribute__((ext_vector_type(8)))  float  v8f;
typedef __attribute__((ext_vector_type(4)))  unsigned int u32x4;
typedef __attribute__((ext_vector_type(8)))  int i32x8;
typedef __attribute__((ext_vector_type(4)))  int i32x4;

#define B_    8
#define C_    256
#define HID_  32
#define N_    4096
#define JB    64              // keys per block (two K=32 chunks for the PV gemm)
#define WAVES 8
#define THREADS (WAVES * 32)
#define VCO   16

// --- toolchain probes -------------------------------------------------------
#if defined(__has_builtin)
#if __has_builtin(__builtin_amdgcn_tensor_load_to_lds)
#define HAVE_TDM 1
#endif
#if __has_builtin(__builtin_amdgcn_ds_swizzle)
#define HAVE_SWZ 1
#endif
#endif
#if defined(__has_include)
#if __has_include(<hip/amd_detail/amd_gfx1250_TDM.h>)
#define TDM_SIX_ARGS 1      // therock-10.0 headers -> 6-arg builtin form
#endif
#endif

// lane-XOR reduction step (immediate-encoded ds_swizzle; falls back to shfl)
template <int PAT>
__device__ __forceinline__ float lxor(float v) {
#if defined(HAVE_SWZ)
  return __int_as_float(__builtin_amdgcn_ds_swizzle(__float_as_int(v), PAT));
#else
  return __shfl_xor(v, (PAT >> 10) & 31, 32);
#endif
}
#define XOR1 0x041F
#define XOR2 0x081F
#define XOR4 0x101F
#define XOR8 0x201F

// pack two f32 into a bf16 pair (lo | hi<<16)
__device__ __forceinline__ unsigned pack_bf16(float a, float b) {
  unsigned short ua = __builtin_bit_cast(unsigned short, (__bf16)a);
  unsigned short ub = __builtin_bit_cast(unsigned short, (__bf16)b);
  return (unsigned)ua | ((unsigned)ub << 16);
}

#if defined(HAVE_TDM)
// low 32 bits of a flat shared-aperture address == LDS byte offset
__device__ __forceinline__ unsigned lds_off(const void* p) {
  return (unsigned)(unsigned long long)(size_t)p;
}
// Issue a 2-D TDM tile load (bf16 elements). D# per CDNA5 ISA 8.3/8.4:
//   group0: count=1 | lds_addr | global_addr(57b) | type=2
//   group1: data_size=1(2B) | tensor_dim0/1 | tile_dim0/1 | dim0_stride
__device__ __forceinline__ void tdm_load_2d(
    unsigned long long gaddr, unsigned ldsaddr,
    unsigned tile_d0, unsigned tile_d1,
    unsigned tensor_d0, unsigned tensor_d1, unsigned row_stride) {
  u32x4 g0;
  g0[0] = 1u;                                          // count=1 (valid)
  g0[1] = ldsaddr;                                     // lds_addr
  g0[2] = (unsigned)(gaddr & 0xFFFFFFFFull);           // global_addr[31:0]
  g0[3] = (unsigned)((gaddr >> 32) & 0x1FFFFFFull)     // global_addr[56:32]
        | (2u << 30);                                  // type=2 ("image")
  i32x8 g1;
  g1[0] = (int)(1u << 16);                             // data_size=1 (2 bytes)
  g1[1] = (int)((tensor_d0 & 0xFFFFu) << 16);          // tensor_dim0[15:0]
  g1[2] = (int)(((tensor_d0 >> 16) & 0xFFFFu) | ((tensor_d1 & 0xFFFFu) << 16));
  g1[3] = (int)(((tensor_d1 >> 16) & 0xFFFFu) | ((tile_d0 & 0xFFFFu) << 16));
  g1[4] = (int)(tile_d1 & 0xFFFFu);                    // tile_dim1 (tile_dim2=0)
  g1[5] = (int)row_stride;                             // tensor_dim0_stride[31:0]
  g1[6] = 0;
  g1[7] = 0;
  i32x4 z4 = {};
#if defined(TDM_SIX_ARGS)
  i32x8 z8 = {};
  __builtin_amdgcn_tensor_load_to_lds(g0, g1, z4, z4, z8, 0);
#else
  __builtin_amdgcn_tensor_load_to_lds(g0, g1, z4, z4, 0);
#endif
}
#define NBUF 2
#else
#define NBUF 1
#endif

// ---------------------------------------------------------------------------
// q/k projection -> bf16 [B][N][HID] (rows 64B contiguous: WMMA A/B feed)
// ---------------------------------------------------------------------------
__global__ __launch_bounds__(256) void qk_proj_kernel(
    const float* __restrict__ x,
    const float* __restrict__ wq, const float* __restrict__ bq,
    const float* __restrict__ wk, const float* __restrict__ bk,
    __bf16* __restrict__ qout, __bf16* __restrict__ kout) {
  __shared__ float sw[HID_ * C_];
  const int b = blockIdx.y;
  const int i = blockIdx.x * 256 + threadIdx.x;
  const float* xb = x + (size_t)b * C_ * N_;

  for (int phase = 0; phase < 2; ++phase) {
    const float* w    = phase ? wk : wq;
    const float* bias = phase ? bk : bq;
    __bf16* outp      = phase ? kout : qout;
    __syncthreads();
    for (int t = threadIdx.x; t < HID_ * C_; t += 256) sw[t] = w[t];
    __syncthreads();
    float acc[HID_];
#pragma unroll
    for (int o = 0; o < HID_; ++o) acc[o] = bias[o];
    for (int c = 0; c < C_; ++c) {
      float xv = xb[(size_t)c * N_ + i];
#pragma unroll
      for (int o = 0; o < HID_; ++o) acc[o] = fmaf(sw[o * C_ + c], xv, acc[o]);
    }
    __bf16* dst = outp + ((size_t)b * N_ + i) * HID_;
#pragma unroll
    for (int o = 0; o < HID_; ++o) dst[o] = (__bf16)acc[o];
  }
}

// ---------------------------------------------------------------------------
// v projection -> bf16 [B][C][N] (rows contiguous in j for the PV B-matrix)
// ---------------------------------------------------------------------------
__global__ __launch_bounds__(256) void v_proj_kernel(
    const float* __restrict__ x,
    const float* __restrict__ wvp, const float* __restrict__ bvp,
    __bf16* __restrict__ vout) {
  __shared__ float sw[VCO * C_];
  const int b  = blockIdx.z;
  const int co = blockIdx.y * VCO;
  const int i  = blockIdx.x * 256 + threadIdx.x;
  for (int t = threadIdx.x; t < VCO * C_; t += 256)
    sw[t] = wvp[(size_t)co * C_ + t];
  __syncthreads();
  float acc[VCO];
#pragma unroll
  for (int p = 0; p < VCO; ++p) acc[p] = bvp[co + p];
  const float* xb = x + (size_t)b * C_ * N_;
  for (int c = 0; c < C_; ++c) {
    float xv = xb[(size_t)c * N_ + i];
#pragma unroll
    for (int p = 0; p < VCO; ++p) acc[p] = fmaf(sw[p * C_ + c], xv, acc[p]);
  }
#pragma unroll
  for (int p = 0; p < VCO; ++p)
    vout[((size_t)b * C_ + co + p) * N_ + i] = (__bf16)acc[p];
}

// ---------------------------------------------------------------------------
// Flash attention: 8 waves/WG; wave = 16 queries x 256 channels; 64-key
// blocks double-buffered via Tensor Data Mover. Per block per wave:
//   4 score WMMAs + 2 rowsum WMMAs (P x ones) + 32 PV WMMAs.
// Key tiling within each 32-chunk is interleaved (tile t covers keys 2n+t) so
// each lane's two P values are column-adjacent -> packed b32 LDS stores.
// ---------------------------------------------------------------------------
__global__ __launch_bounds__(THREADS) void attn_kernel(
    const __bf16* __restrict__ qg,   // [B][N][HID]
    const __bf16* __restrict__ kg,   // [B][N][HID]
    const __bf16* __restrict__ vg,   // [B][C][N]
    float* __restrict__ out) {       // [B][C][N]
  __shared__ __align__(16) __bf16 sK[NBUF][JB * HID_];   // [j][d]  4 KB each
  __shared__ __align__(16) __bf16 sV[NBUF][C_ * JB];     // [c][j] 32 KB each
  __shared__ __align__(16) __bf16 sP[WAVES][16 * JB];    // wave [i][j] 2 KB each

  const int b    = blockIdx.y;
  const int tid  = threadIdx.x;
  const int wid  = tid >> 5;
  const int lane = tid & 31;
  const int lh   = lane >> 4;
  const int l16  = lane & 15;
  const int i0   = (blockIdx.x * WAVES + wid) * 16;

  const int adb = lh ? 8 : 0;    // A-layout K-base (lanes 0-15: K 0-7 / 16-23)
  const int bdb = lh ? 8 : 0;    // B-layout K-base (lanes 0-15: K 0-7 / 16-23)

  // ---- preload Q tile as WMMA A-matrix (16 x 32 bf16) ----
  const __bf16* qrow = qg + ((size_t)b * N_ + i0 + l16) * HID_;
  v8bf qlo = *(const v8bf*)(qrow + adb);
  v8bf qhi = *(const v8bf*)(qrow + adb + 16);
  v16bf aQ;
#pragma unroll
  for (int e = 0; e < 8; ++e) { aQ[e] = qlo[e]; aQ[8 + e] = qhi[e]; }

  // all-ones B matrix for rowsum-by-WMMA
  v16bf ones;
#pragma unroll
  for (int e = 0; e < 16; ++e) ones[e] = (__bf16)1.0f;

  v8f acc[C_ / 16];
  v8f zero8 = {};
#pragma unroll
  for (int t = 0; t < C_ / 16; ++t) acc[t] = zero8;
  float runmax[8], runsum[8];
#pragma unroll
  for (int r = 0; r < 8; ++r) { runmax[r] = -3.0e38f; runsum[r] = 0.0f; }

  __bf16* myP = sP[wid];
  const int NB = N_ / JB;

#if defined(HAVE_TDM)
  // prologue: DMA block 0 into buffer 0
  if (wid == 0) {
    tdm_load_2d((unsigned long long)(size_t)(kg + ((size_t)b * N_) * HID_),
                lds_off(&sK[0][0]), HID_, JB, HID_, N_, HID_);
    tdm_load_2d((unsigned long long)(size_t)(vg + (size_t)b * C_ * N_),
                lds_off(&sV[0][0]), JB, C_, N_, C_, N_);
  }
  int cur = 0;
#endif

  for (int kb = 0; kb < NB; ++kb) {
    const int j0 = kb * JB;
#if defined(HAVE_TDM)
    if (wid == 0) __builtin_amdgcn_s_wait_tensorcnt(0);
    __syncthreads();                       // buf[cur] ready; block kb-1 done
    if (wid == 0 && kb + 1 < NB) {         // DMA next block into other buffer
      const int jn = j0 + JB;
      tdm_load_2d((unsigned long long)(size_t)(kg + ((size_t)b * N_ + jn) * HID_),
                  lds_off(&sK[cur ^ 1][0]), HID_, JB, HID_, N_, HID_);
      tdm_load_2d((unsigned long long)(size_t)(vg + (size_t)b * C_ * N_ + jn),
                  lds_off(&sV[cur ^ 1][0]), JB, C_, N_, C_, N_);
    }
    const __bf16* kbp = sK[cur];
    const __bf16* vbp = sV[cur];
#else
    __syncthreads();
    // manual staging fallback: K 256 chunks, V 2048 chunks of 16B
    {
      int row = tid >> 2, off = (tid & 3) * 8;
      *(v8bf*)(&sK[0][row * HID_ + off]) =
          *(const v8bf*)(kg + ((size_t)b * N_ + j0 + row) * HID_ + off);
    }
#pragma unroll
    for (int rep = 0; rep < 8; ++rep) {
      int chunk = tid + rep * THREADS;
      int row = chunk >> 3, off = (chunk & 7) * 8;
      *(v8bf*)(&sV[0][row * JB + off]) =
          *(const v8bf*)(vg + ((size_t)b * C_ + row) * N_ + j0 + off);
    }
    if (j0 + JB < N_) {
      __builtin_prefetch(kg + ((size_t)b * N_ + j0 + JB + (tid >> 2)) * HID_, 0, 0);
      __builtin_prefetch(vg + ((size_t)b * C_ + tid) * N_ + j0 + JB, 0, 0);
    }
    __syncthreads();
    const __bf16* kbp = sK[0];
    const __bf16* vbp = sV[0];
#endif

    // ---- scores: 4 WMMAs; chunk q, tile t covers keys q*32 + 2n + t ----
    v8f S[4];
#pragma unroll
    for (int q = 0; q < 2; ++q)
#pragma unroll
      for (int t = 0; t < 2; ++t) {
        const __bf16* r1 = kbp + (q * 32 + 2 * l16 + t) * HID_ + bdb;
        v8bf x0 = *(const v8bf*)r1, x1 = *(const v8bf*)(r1 + 16);
        v16bf bK;
#pragma unroll
        for (int e = 0; e < 8; ++e) { bK[e] = x0[e]; bK[8 + e] = x1[e]; }
        S[q * 2 + t] = __builtin_amdgcn_wmma_f32_16x16x32_bf16(
            false, aQ, false, bK, (short)0, zero8, false, false);
      }

    // ---- online max + exp; packed P stores (two bf16 per b32) ----
    float scale[8];
#pragma unroll
    for (int r = 0; r < 8; ++r) {
      float mx = fmaxf(fmaxf(S[0][r], S[1][r]), fmaxf(S[2][r], S[3][r]));
      mx = fmaxf(mx, lxor<XOR1>(mx));
      mx = fmaxf(mx, lxor<XOR2>(mx));
      mx = fmaxf(mx, lxor<XOR4>(mx));
      mx = fmaxf(mx, lxor<XOR8>(mx));
      float nm = fmaxf(runmax[r], mx);
      scale[r] = __expf(runmax[r] - nm);
      runmax[r] = nm;
      float p00 = __expf(S[0][r] - nm);
      float p01 = __expf(S[1][r] - nm);
      float p10 = __expf(S[2][r] - nm);
      float p11 = __expf(S[3][r] - nm);
      int m = lh ? (8 + r) : r;
      *(unsigned*)(myP + m * JB + 2 * l16)      = pack_bf16(p00, p01);
      *(unsigned*)(myP + m * JB + 32 + 2 * l16) = pack_bf16(p10, p11);
    }
    // rescale running output accumulators
#pragma unroll
    for (int t = 0; t < C_ / 16; ++t)
#pragma unroll
      for (int r = 0; r < 8; ++r) acc[t][r] *= scale[r];

    // ---- reload P as two A-matrices (16 x 32 each) ----
    v16bf aP[2];
#pragma unroll
    for (int q = 0; q < 2; ++q) {
      const __bf16* pr = myP + l16 * JB + q * 32 + adb;
      v8bf p0 = *(const v8bf*)pr, p1 = *(const v8bf*)(pr + 16);
#pragma unroll
      for (int e = 0; e < 8; ++e) { aP[q][e] = p0[e]; aP[q][8 + e] = p1[e]; }
    }

    // ---- block row sums via WMMA (P x ones): every lane gets its 8 rows ----
    v8f rs = __builtin_amdgcn_wmma_f32_16x16x32_bf16(
        false, aP[0], false, ones, (short)0, zero8, false, false);
    rs = __builtin_amdgcn_wmma_f32_16x16x32_bf16(
        false, aP[1], false, ones, (short)0, rs, false, false);
#pragma unroll
    for (int r = 0; r < 8; ++r) runsum[r] = runsum[r] * scale[r] + rs[r];

    // ---- PV: two K=32 chunks x 16 channel tiles = 32 WMMAs ----
#pragma unroll
    for (int q = 0; q < 2; ++q)
#pragma unroll
      for (int ct = 0; ct < C_ / 16; ++ct) {
        const __bf16* vr = vbp + (ct * 16 + l16) * JB + q * 32 + bdb;
        v8bf v0 = *(const v8bf*)vr, v1 = *(const v8bf*)(vr + 16);
        v16bf bV;
#pragma unroll
        for (int e = 0; e < 8; ++e) { bV[e] = v0[e]; bV[8 + e] = v1[e]; }
        acc[ct] = __builtin_amdgcn_wmma_f32_16x16x32_bf16(
            false, aP[q], false, bV, (short)0, acc[ct], false, false);
      }
#if defined(HAVE_TDM)
    cur ^= 1;
#endif
  }

  // ---- epilogue: normalize, write [B][C][N] f32 ----
  float inv[8];
#pragma unroll
  for (int r = 0; r < 8; ++r) inv[r] = 1.0f / runsum[r];
#pragma unroll
  for (int t = 0; t < C_ / 16; ++t) {
    int c = t * 16 + l16;
    float* op = out + ((size_t)b * C_ + c) * N_ + i0;
#pragma unroll
    for (int r = 0; r < 8; ++r) {
      int m = lh ? (8 + r) : r;
      op[m] = acc[t][r] * inv[r];
    }
  }
}

// ---------------------------------------------------------------------------
extern "C" void kernel_launch(void* const* d_in, const int* in_sizes, int n_in,
                              void* d_out, int out_size, void* d_ws, size_t ws_size,
                              hipStream_t stream) {
  const float* x  = (const float*)d_in[0];
  const float* wq = (const float*)d_in[1];
  const float* bq = (const float*)d_in[2];
  const float* wk = (const float*)d_in[3];
  const float* bk = (const float*)d_in[4];
  const float* wv = (const float*)d_in[5];
  const float* bv = (const float*)d_in[6];
  float* out = (float*)d_out;

  // workspace: q (2MB) | k (2MB) | v (16MB), all bf16
  __bf16* qg = (__bf16*)d_ws;
  __bf16* kg = qg + (size_t)B_ * N_ * HID_;
  __bf16* vg = kg + (size_t)B_ * N_ * HID_;

  qk_proj_kernel<<<dim3(N_ / 256, B_), 256, 0, stream>>>(x, wq, bq, wk, bk, qg, kg);
  v_proj_kernel<<<dim3(N_ / 256, C_ / VCO, B_), 256, 0, stream>>>(x, wv, bv, vg);
  attn_kernel<<<dim3(N_ / (WAVES * 16), B_), THREADS, 0, stream>>>(qg, kg, vg, out);
}
